// MerelModel_86354612453923
// MI455X (gfx1250) — matrix operand order repaired and checked by reading, not compiled, not run on hardware
//
#include <hip/hip_runtime.h>
#include <hip/hip_bf16.h>
#include <stdint.h>

// ---------------------------------------------------------------------------
// Types for WMMA
// ---------------------------------------------------------------------------
typedef __attribute__((ext_vector_type(16))) _Float16 v16h;
typedef __attribute__((ext_vector_type(8)))  _Float16 v8h;
typedef __attribute__((ext_vector_type(4)))  _Float16 v4h;
typedef __attribute__((ext_vector_type(8)))  float    v8f;

#define Tn 128
#define Bn 16
#define TB 2048            // T*B
#define HH 256             // core hidden
#define GATES 1024         // 4*HH
#define CIN 384            // core input dim
#define PIN 748            // policy input dim
#define PA 38              // policy hidden (ACT)
#define PG 152             // 4*PA

__device__ __forceinline__ float sigmf(float x) { return 1.0f / (1.0f + __expf(-x)); }

// ---------------------------------------------------------------------------
// Fragment loaders (layouts per CDNA5 ISA 7.12.2, wave32)
// ---------------------------------------------------------------------------
// A fragment 16x32 f16 from LDS (row-major, `stride` halves per row).
// lane<16: row=lane,    K = {k0+0..7,  k0+16..23}
// lane>=16: row=lane-16, K = {k0+8..15, k0+24..31}
__device__ __forceinline__ v16h frag_a_lds(const _Float16* base, int stride,
                                           int m0, int k0, int lid) {
  const _Float16* p = base + (m0 + (lid & 15)) * stride + k0 + ((lid < 16) ? 0 : 8);
  v16h r;
  *(v8h*)&r           = *(const v8h*)p;         // K run 0
  *(((v8h*)&r) + 1)   = *(const v8h*)(p + 16);  // K run +16
  return r;
}

// B fragment 32x16 f16 from LDS stored TRANSPOSED as [n][k] rows.
// lane<16: n=lane, K = {k0+0..7, k0+16..23} ; lane>=16: n=lane-16, K = {k0+8..15, k0+24..31}
__device__ __forceinline__ v16h frag_b_lds(const _Float16* baseT, int stride,
                                           int n0, int k0, int lid) {
  const _Float16* p = baseT + (n0 + (lid & 15)) * stride + k0 + ((lid < 16) ? 0 : 8);
  v16h r;
  *(v8h*)&r         = *(const v8h*)p;
  *(((v8h*)&r) + 1) = *(const v8h*)(p + 16);
  return r;
}

// A fragment built directly from f32 global weights (row-major, ld floats/row)
__device__ __forceinline__ v16h frag_a_g32(const float* W, int ld,
                                           int m0, int k0, int lid) {
  const float* p = W + (size_t)(m0 + (lid & 15)) * ld + k0 + ((lid < 16) ? 0 : 8);
  v16h r;
#pragma unroll
  for (int i = 0; i < 8; ++i) {
    r[i]     = (_Float16)p[i];
    r[i + 8] = (_Float16)p[i + 16];
  }
  return r;
}

// ---------------------------------------------------------------------------
// Generic WMMA GEMM:  C(MxN) = act( A(MxK) * B(KxN) + bias1 + bias2 )
// A,B f32 global (B optionally given as (N,K) row-major -> transB=1),
// converted to f16 while staging into LDS. f32 accumulate via WMMA.
// Block tile 128x64, KC=64 (2 WMMA K-steps per barrier), 256 threads =
// 8 waves (4x2), wave tile 32x32. Vectorized b128 staging fast path when
// the chunk is fully in-bounds (uniform predicate, EXEC stays full).
// ---------------------------------------------------------------------------
#define BM 128
#define BN 64
#define KC 64
#define SA_STRIDE 72
#define SB_STRIDE 72

__global__ __launch_bounds__(256) void gemm_wmma_f16(
    const float* __restrict__ A, int lda,
    const float* __restrict__ B, int ldb, int transB,
    const float* __restrict__ bias1, const float* __restrict__ bias2,
    float* __restrict__ C, int ldc,
    int M, int N, int K, int doRelu) {
  __shared__ _Float16 sA[BM * SA_STRIDE];
  __shared__ _Float16 sB[BN * SB_STRIDE];   // transposed: [n][k]

  const int tid  = threadIdx.x;
  const int lid  = tid & 31;
  const int wave = tid >> 5;
  const int wm   = (wave >> 1) * 32;   // wave M offset in tile
  const int wn   = (wave & 1) * 32;    // wave N offset in tile
  const int bm   = blockIdx.y * BM;
  const int bn   = blockIdx.x * BN;

  v8f acc[2][2];
#pragma unroll
  for (int i = 0; i < 2; ++i)
#pragma unroll
    for (int j = 0; j < 2; ++j) acc[i][j] = (v8f){};

  const int nchunks = (K + KC - 1) / KC;
  for (int ck = 0; ck < nchunks; ++ck) {
    const int kb = ck * KC;

    // ---- stage A chunk (BM x KC) f32 -> f16 ----
    const bool fullA = (bm + BM <= M) && (kb + KC <= K);
    if (fullA) {
      const float* Abase = A + (size_t)bm * lda + kb;
      for (int s = tid; s < BM * (KC / 4); s += 256) {   // 2048 float4 slots
        int r  = s >> 4;               // KC/4 == 16 slots per row
        int c4 = (s & 15) << 2;
        float4 v = *(const float4*)(Abase + (size_t)r * lda + c4);
        v4h h = { (_Float16)v.x, (_Float16)v.y, (_Float16)v.z, (_Float16)v.w };
        *(v4h*)(sA + r * SA_STRIDE + c4) = h;            // ds_store_b64
      }
    } else {
      for (int i = tid; i < BM * KC; i += 256) {
        int r = i >> 6, c = i & 63;
        int gr = bm + r, gc = kb + c;
        float v = (gr < M && gc < K) ? A[(size_t)gr * lda + gc] : 0.0f;
        sA[r * SA_STRIDE + c] = (_Float16)v;
      }
    }

    // ---- stage B chunk transposed (BN x KC as [n][k]) ----
    const bool fullB = (bn + BN <= N) && (kb + KC <= K);
    if (fullB) {
      if (transB) {
        for (int s = tid; s < BN * (KC / 4); s += 256) { // 1024 float4 slots
          int n = s >> 4, c4 = (s & 15) << 2;
          float4 v = *(const float4*)(B + (size_t)(bn + n) * ldb + kb + c4);
          v4h h = { (_Float16)v.x, (_Float16)v.y, (_Float16)v.z, (_Float16)v.w };
          *(v4h*)(sB + n * SB_STRIDE + c4) = h;
        }
      } else {
        for (int s = tid; s < KC * (BN / 4); s += 256) { // 1024 float4 slots
          int k = s >> 4, n4 = (s & 15) << 2;
          float4 v = *(const float4*)(B + (size_t)(kb + k) * ldb + bn + n4);
          sB[(n4 + 0) * SB_STRIDE + k] = (_Float16)v.x;
          sB[(n4 + 1) * SB_STRIDE + k] = (_Float16)v.y;
          sB[(n4 + 2) * SB_STRIDE + k] = (_Float16)v.z;
          sB[(n4 + 3) * SB_STRIDE + k] = (_Float16)v.w;
        }
      }
    } else {
      for (int i = tid; i < BN * KC; i += 256) {
        int n = i >> 6, c = i & 63;
        int gn = bn + n, gc = kb + c;
        float v = 0.0f;
        if (gn < N && gc < K)
          v = transB ? B[(size_t)gn * ldb + gc] : B[(size_t)gc * ldb + gn];
        sB[n * SB_STRIDE + c] = (_Float16)v;
      }
    }

    // prefetch next K chunk (global_prefetch_b8)
    if (ck + 1 < nchunks) {
      int kn = kb + KC;
      if (tid < BM) {
        int gr = bm + tid;
        if (gr < M && kn < K) __builtin_prefetch(&A[(size_t)gr * lda + kn], 0, 1);
      } else if (tid < BM + BN) {
        int gn = bn + (tid - BM);
        if (gn < N && kn < K) {
          const float* p = transB ? &B[(size_t)gn * ldb + kn] : &B[(size_t)kn * ldb + gn];
          __builtin_prefetch(p, 0, 1);
        }
      }
    }
    __syncthreads();

#pragma unroll
    for (int ks = 0; ks < KC; ks += 32) {
      v16h a0 = frag_a_lds(sA, SA_STRIDE, wm,      ks, lid);
      v16h a1 = frag_a_lds(sA, SA_STRIDE, wm + 16, ks, lid);
      v16h b0 = frag_b_lds(sB, SB_STRIDE, wn,      ks, lid);
      v16h b1 = frag_b_lds(sB, SB_STRIDE, wn + 16, ks, lid);
      acc[0][0] = __builtin_amdgcn_wmma_f32_16x16x32_f16(false, a0, false, b0, (short)0, acc[0][0], false, false);
      acc[0][1] = __builtin_amdgcn_wmma_f32_16x16x32_f16(false, a0, false, b1, (short)0, acc[0][1], false, false);
      acc[1][0] = __builtin_amdgcn_wmma_f32_16x16x32_f16(false, a1, false, b0, (short)0, acc[1][0], false, false);
      acc[1][1] = __builtin_amdgcn_wmma_f32_16x16x32_f16(false, a1, false, b1, (short)0, acc[1][1], false, false);
    }
    __syncthreads();
  }

  // ---- epilogue: D lane layout -> global f32 ----
  const int half = lid >> 4;
#pragma unroll
  for (int i = 0; i < 2; ++i) {
#pragma unroll
    for (int j = 0; j < 2; ++j) {
      int col = bn + wn + j * 16 + (lid & 15);
      float bv = 0.0f;
      if (col < N) {
        if (bias1) bv += bias1[col];
        if (bias2) bv += bias2[col];
      }
#pragma unroll
      for (int r = 0; r < 8; ++r) {
        int row = bm + wm + i * 16 + r + half * 8;
        if (row < M && col < N) {
          float v = acc[i][j][r] + bv;
          if (doRelu) v = fmaxf(v, 0.0f);
          C[(size_t)row * ldc + col] = v;
        }
      }
    }
  }
}

// ---------------------------------------------------------------------------
// Core LSTM scan (H=256, B=16, T=128). Single persistent workgroup, 32 waves.
// x@Wih^T (+biases) precomputed in gates_x. Per step: gates_h = Whh * h^T via
// WMMA (A = Whh 16x32 f16 fragments, register-resident; B = h^T from LDS),
// then fused elementwise LSTM cell with episode-reset masking.
// ---------------------------------------------------------------------------
#define HB_STRIDE 264      // halves per batch row of h buffer
#define GB_STRIDE 1040     // halves per batch row of gate buffer

__global__ __launch_bounds__(1024) void core_lstm_kernel(
    const float* __restrict__ gates_x,   // (TB, 1024) incl. bih+bhh
    const float* __restrict__ Whh,       // (1024, 256) f32 row-major
    const uint8_t* __restrict__ done,    // (B, T)
    float* __restrict__ hout) {          // (TB, 256)
  __shared__ _Float16 hbuf[Bn * HB_STRIDE];   // h^T, [batch][256] f16 (B-frag layout)
  __shared__ _Float16 gbuf[Bn * GB_STRIDE];   // gates_h, [batch][1024] f16

  const int tid  = threadIdx.x;
  const int lid  = tid & 31;
  const int wave = tid >> 5;
  const int m0   = wave * 32;          // this wave's 32 gate rows

  // Preload Whh WMMA A-fragments (register resident): 2 M-tiles x 8 K-chunks
  v16h wfrag[2][8];
#pragma unroll
  for (int i = 0; i < 2; ++i)
#pragma unroll
    for (int kc = 0; kc < 8; ++kc)
      wfrag[i][kc] = frag_a_g32(Whh, HH, m0 + i * 16, kc * 32, lid);

  // Per-thread recurrent state: unit u, batches b0..b0+3
  const int u  = tid & 255;
  const int b0 = (tid >> 8) * 4;
  float hreg[4] = {0.f, 0.f, 0.f, 0.f};
  float creg[4] = {0.f, 0.f, 0.f, 0.f};

  for (int i = tid; i < Bn * HB_STRIDE; i += 1024) hbuf[i] = (_Float16)0.0f;
  __syncthreads();

  for (int t = 0; t < Tn; ++t) {
    // phase 0: apply episode reset, publish h to LDS
#pragma unroll
    for (int q = 0; q < 4; ++q) {
      int b = b0 + q;
      float keep = done[b * Tn + t] ? 0.0f : 1.0f;
      hreg[q] *= keep;
      creg[q] *= keep;
      hbuf[b * HB_STRIDE + u] = (_Float16)hreg[q];
    }
    __syncthreads();

    // phase 1: gates_h = Whh * h^T  (16x16x32 WMMA, N = batch)
    v16h hf[8];
#pragma unroll
    for (int kc = 0; kc < 8; ++kc)
      hf[kc] = frag_b_lds(hbuf, HB_STRIDE, 0, kc * 32, lid);

    const int bcol = lid & 15;
    const int halfv = lid >> 4;
#pragma unroll
    for (int i = 0; i < 2; ++i) {
      v8f acc = (v8f){};
#pragma unroll
      for (int kc = 0; kc < 8; ++kc)
        acc = __builtin_amdgcn_wmma_f32_16x16x32_f16(false, wfrag[i][kc], false, hf[kc],
                                                     (short)0, acc, false, false);
#pragma unroll
      for (int r = 0; r < 8; ++r)
        gbuf[bcol * GB_STRIDE + (m0 + i * 16 + r + halfv * 8)] = (_Float16)acc[r];
    }
    __syncthreads();

    // phase 2: LSTM cell (torch gate order i,f,g,o)
#pragma unroll
    for (int q = 0; q < 4; ++q) {
      int b = b0 + q;
      const float* gx = gates_x + (size_t)(t * Bn + b) * GATES;
      const _Float16* gh = gbuf + b * GB_STRIDE;
      float gi = (float)gh[u]       + gx[u];
      float gf = (float)gh[u + 256] + gx[u + 256];
      float gg = (float)gh[u + 512] + gx[u + 512];
      float go = (float)gh[u + 768] + gx[u + 768];
      float c = sigmf(gf) * creg[q] + sigmf(gi) * tanhf(gg);
      float h = sigmf(go) * tanhf(c);
      creg[q] = c;
      hreg[q] = h;
      hout[(size_t)(t * Bn + b) * HH + u] = h;
    }
    __syncthreads();
  }
}

// ---------------------------------------------------------------------------
// Policy 3-layer LSTM, H=38 (too small for WMMA tiles) — pure VALU scan,
// single workgroup. Layer0 x-projection precomputed by WMMA GEMM (pgx).
// ---------------------------------------------------------------------------
__global__ __launch_bounds__(256) void policy_lstm_kernel(
    const float* __restrict__ pgx,                        // (TB, 152) incl. biases
    const float* __restrict__ W1ih, const float* __restrict__ W2ih,   // (152,38)
    const float* __restrict__ Whh0, const float* __restrict__ Whh1,
    const float* __restrict__ Whh2,                       // (152,38)
    const float* __restrict__ b1ih, const float* __restrict__ b1hh,
    const float* __restrict__ b2ih, const float* __restrict__ b2hh,   // (152)
    const uint8_t* __restrict__ done,                     // (B,T)
    float* __restrict__ xtmp0, float* __restrict__ xtmp1, // (TB, 38)
    float* __restrict__ out_policy) {                     // (TB, 38)
  __shared__ float hs[PA * Bn];
  __shared__ float cs[PA * Bn];
  __shared__ float gs[PG * Bn];
  const int tid = threadIdx.x;

  for (int layer = 0; layer < 3; ++layer) {
    const float* Whh  = (layer == 0) ? Whh0 : ((layer == 1) ? Whh1 : Whh2);
    const float* Wih  = (layer == 1) ? W1ih : W2ih;
    const float* bih  = (layer == 1) ? b1ih : b2ih;
    const float* bhh  = (layer == 1) ? b1hh : b2hh;
    const float* xin  = (layer == 1) ? xtmp0 : xtmp1;
    float* xout = (layer == 0) ? xtmp0 : ((layer == 1) ? xtmp1 : out_policy);

    for (int i = tid; i < PA * Bn; i += 256) { hs[i] = 0.0f; cs[i] = 0.0f; }
    __syncthreads();

    for (int t = 0; t < Tn; ++t) {
      // reset mask
      for (int e = tid; e < PA * Bn; e += 256) {
        int b = e & 15;
        float keep = done[b * Tn + t] ? 0.0f : 1.0f;
        hs[e] *= keep;
        cs[e] *= keep;
      }
      __syncthreads();
      // gates = x-part + Whh @ h
      for (int e = tid; e < PG * Bn; e += 256) {
        int g = e >> 4, b = e & 15;
        float a;
        if (layer == 0) {
          a = pgx[(size_t)(t * Bn + b) * PG + g];
        } else {
          a = bih[g] + bhh[g];
          const float* xr = xin + (size_t)(t * Bn + b) * PA;
          for (int k = 0; k < PA; ++k) a += Wih[g * PA + k] * xr[k];
        }
        for (int k = 0; k < PA; ++k) a += Whh[g * PA + k] * hs[k * Bn + b];
        gs[g * Bn + b] = a;
      }
      __syncthreads();
      // cell
      for (int e = tid; e < PA * Bn; e += 256) {
        int uu = e >> 4, b = e & 15;
        float gi = gs[uu * Bn + b];
        float gf = gs[(uu + PA) * Bn + b];
        float gg = gs[(uu + 2 * PA) * Bn + b];
        float go = gs[(uu + 3 * PA) * Bn + b];
        float c = sigmf(gf) * cs[e] + sigmf(gi) * tanhf(gg);
        float h = sigmf(go) * tanhf(c);
        cs[e] = c;
        hs[e] = h;
        xout[(size_t)(t * Bn + b) * PA + uu] = h;
      }
      __syncthreads();
    }
    __syncthreads();
  }
}

// ---------------------------------------------------------------------------
// pol_in = [core_h | ft | propri]  (TB x 748)
// ---------------------------------------------------------------------------
__global__ void concat_kernel(const float* __restrict__ core_h,
                              const float* __restrict__ ft,
                              const float* __restrict__ propri,
                              float* __restrict__ pol_in) {
  int idx = blockIdx.x * blockDim.x + threadIdx.x;
  if (idx >= TB * PIN) return;
  int r = idx / PIN, c = idx % PIN;
  float v;
  if (c < HH)            v = core_h[(size_t)r * HH + c];
  else if (c < HH + CIN) v = ft[(size_t)r * CIN + (c - HH)];
  else                   v = propri[(size_t)r * 108 + (c - HH - CIN)];
  pol_in[idx] = v;
}

// ---------------------------------------------------------------------------
// value = v4 @ outW + outb   (2048 rows x K=256 -> 1)
// ---------------------------------------------------------------------------
__global__ void value_out_kernel(const float* __restrict__ v,
                                 const float* __restrict__ outW,
                                 const float* __restrict__ outb,
                                 float* __restrict__ out) {
  int r = blockIdx.x * blockDim.x + threadIdx.x;
  if (r >= TB) return;
  float acc = outb[0];
  const float* vr = v + (size_t)r * HH;
  for (int k = 0; k < HH; ++k) acc += vr[k] * outW[k];
  out[r] = acc;
}

// ---------------------------------------------------------------------------
// Host-side launcher
// ---------------------------------------------------------------------------
extern "C" void kernel_launch(void* const* d_in, const int* in_sizes, int n_in,
                              void* d_out, int out_size, void* d_ws, size_t ws_size,
                              hipStream_t stream) {
  (void)in_sizes; (void)n_in; (void)out_size; (void)ws_size;
  const float*   visual   = (const float*)d_in[0];
  const float*   propri   = (const float*)d_in[1];
  const uint8_t* done     = (const uint8_t*)d_in[2];
  const float*   vW1      = (const float*)d_in[3];
  const float*   vb1      = (const float*)d_in[4];
  const float*   vW2      = (const float*)d_in[5];
  const float*   vb2      = (const float*)d_in[6];
  const float*   pW1      = (const float*)d_in[7];
  const float*   pb1      = (const float*)d_in[8];
  const float*   pW2      = (const float*)d_in[9];
  const float*   pb2      = (const float*)d_in[10];
  const float*   core_Wih = (const float*)d_in[11];
  const float*   core_Whh = (const float*)d_in[12];
  const float*   core_bih = (const float*)d_in[13];
  const float*   core_bhh = (const float*)d_in[14];
  const float*   pol0_Wih = (const float*)d_in[15];
  const float*   pol0_Whh = (const float*)d_in[16];
  const float*   pol0_bih = (const float*)d_in[17];
  const float*   pol0_bhh = (const float*)d_in[18];
  const float*   pol1_Wih = (const float*)d_in[19];
  const float*   pol1_Whh = (const float*)d_in[20];
  const float*   pol1_bih = (const float*)d_in[21];
  const float*   pol1_bhh = (const float*)d_in[22];
  const float*   pol2_Wih = (const float*)d_in[23];
  const float*   pol2_Whh = (const float*)d_in[24];
  const float*   pol2_bih = (const float*)d_in[25];
  const float*   pol2_bhh = (const float*)d_in[26];
  const float*   mlpW     = (const float*)d_in[27];
  const float*   mlpb     = (const float*)d_in[28];
  const float*   outW     = (const float*)d_in[29];
  const float*   outb     = (const float*)d_in[30];

  float* out_value  = (float*)d_out;            // (TB)
  float* out_policy = (float*)d_out + TB;       // (TB, 38)

  // workspace layout (f32 elements)
  float* ws = (float*)d_ws;
  float* vh1     = ws;                         // 2048*256
  float* ft      = vh1 + TB * 256;             // 2048*384
  float* ph1     = ft + TB * CIN;              // 2048*128
  float* gates_x = ph1 + TB * 128;             // 2048*1024
  float* core_h  = gates_x + TB * GATES;       // 2048*256
  float* pol_in  = core_h + TB * HH;           // 2048*748
  float* pgx     = pol_in + TB * PIN;          // 2048*152
  float* xtmp0   = pgx + TB * PG;              // 2048*38
  float* xtmp1   = xtmp0 + TB * PA;            // 2048*38
  float* vping   = xtmp1 + TB * PA;            // 2048*256
  float* vpong   = vping + TB * HH;            // 2048*256

  auto grid = [](int M, int N) { return dim3((N + BN - 1) / BN, (M + BM - 1) / BM); };
  dim3 blk(256);

  // 1. visual encoder layer 1: relu(vflat @ vW1 + vb1)    M=2048 K=12288 N=256
  gemm_wmma_f16<<<grid(TB, 256), blk, 0, stream>>>(visual, 12288, vW1, 256, 0,
      vb1, nullptr, vh1, 256, TB, 256, 12288, 1);
  // 2. visual encoder layer 2 -> ft[:, 0:256]
  gemm_wmma_f16<<<grid(TB, 256), blk, 0, stream>>>(vh1, 256, vW2, 256, 0,
      vb2, nullptr, ft, CIN, TB, 256, 256, 0);
  // 3. propri encoder layer 1: relu(propri @ pW1 + pb1)   K=108 N=128
  gemm_wmma_f16<<<grid(TB, 128), blk, 0, stream>>>(propri, 108, pW1, 128, 0,
      pb1, nullptr, ph1, 128, TB, 128, 108, 1);
  // 4. propri encoder layer 2 -> ft[:, 256:384]
  gemm_wmma_f16<<<grid(TB, 128), blk, 0, stream>>>(ph1, 128, pW2, 128, 0,
      pb2, nullptr, ft + 256, CIN, TB, 128, 128, 0);
  // 5. core input projection: gates_x = ft @ core_Wih^T + bih + bhh   N=1024 K=384
  gemm_wmma_f16<<<grid(TB, GATES), blk, 0, stream>>>(ft, CIN, core_Wih, CIN, 1,
      core_bih, core_bhh, gates_x, GATES, TB, GATES, CIN, 0);
  // 6. core LSTM scan (persistent single workgroup, WMMA recurrent GEMM)
  core_lstm_kernel<<<1, 1024, 0, stream>>>(gates_x, core_Whh, done, core_h);
  // 7. policy input concat
  {
    int total = TB * PIN;
    concat_kernel<<<(total + 255) / 256, 256, 0, stream>>>(core_h, ft, propri, pol_in);
  }
  // 8. policy layer-0 input projection: pgx = pol_in @ pol0_Wih^T + biases  N=152 K=748
  gemm_wmma_f16<<<grid(TB, PG), blk, 0, stream>>>(pol_in, PIN, pol0_Wih, PIN, 1,
      pol0_bih, pol0_bhh, pgx, PG, TB, PG, PIN, 0);
  // 9. policy 3-layer LSTM scan (writes policy output)
  policy_lstm_kernel<<<1, 256, 0, stream>>>(pgx, pol1_Wih, pol2_Wih,
      pol0_Whh, pol1_Whh, pol2_Whh, pol1_bih, pol1_bhh, pol2_bih, pol2_bhh,
      done, xtmp0, xtmp1, out_policy);
  // 10-13. value MLP (4 relu layers on core_h)
  gemm_wmma_f16<<<grid(TB, HH), blk, 0, stream>>>(core_h, HH, mlpW + 0 * HH * HH, HH, 0,
      mlpb + 0 * HH, nullptr, vping, HH, TB, HH, HH, 1);
  gemm_wmma_f16<<<grid(TB, HH), blk, 0, stream>>>(vping, HH, mlpW + 1 * HH * HH, HH, 0,
      mlpb + 1 * HH, nullptr, vpong, HH, TB, HH, HH, 1);
  gemm_wmma_f16<<<grid(TB, HH), blk, 0, stream>>>(vpong, HH, mlpW + 2 * HH * HH, HH, 0,
      mlpb + 2 * HH, nullptr, vping, HH, TB, HH, HH, 1);
  gemm_wmma_f16<<<grid(TB, HH), blk, 0, stream>>>(vping, HH, mlpW + 3 * HH * HH, HH, 0,
      mlpb + 3 * HH, nullptr, vpong, HH, TB, HH, HH, 1);
  // 14. value head
  value_out_kernel<<<(TB + 255) / 256, 256, 0, stream>>>(vpong, outW, outb, out_value);
}
